// RobertaSelfAttention_78675210928833
// MI455X (gfx1250) — compile-verified, matrix operand order checked by reading, not run on hardware
//
#include <hip/hip_runtime.h>
#include <hip/hip_bf16.h>
#include <cstdint>

// Problem constants (RobertaSelfAttention)
#define B_  2
#define S_  4096
#define H_  768
#define NH_ 12
#define HD_ 64
#define M_  (B_ * S_)   // 8192 rows
#define KT_ 128         // keys staged per LDS tile in attention

typedef __attribute__((ext_vector_type(16))) _Float16 v16h;
typedef __attribute__((ext_vector_type(8)))  _Float16 v8h;
typedef __attribute__((ext_vector_type(8)))  float    v8f;

__device__ __forceinline__ v8f wmma16(v16h a, v16h b, v8f c) {
  // emits v_wmma_f32_16x16x32_f16
  return __builtin_amdgcn_wmma_f32_16x16x32_f16(
      /*neg_a=*/false, a, /*neg_b=*/false, b,
      /*c_mod=*/(short)0, c, /*reuse_a=*/false, /*reuse_b=*/false);
}

// Load a 16x32 f16 fragment (16 rows x K=32) from row-major memory (ld in halves).
// CDNA5 16-bit A layout: lane L<16 -> row L, K chunks {0..7} and {16..23};
// lane L>=16 -> row L-16, K chunks {8..15} and {24..31}.
// The same pattern loads B operands from the transposed source (rows = N columns).
__device__ __forceinline__ v16h load_frag(const _Float16* base, int ld, int lane) {
  int row = lane & 15;
  int col = (lane >> 4) << 3;                 // 0 or 8
  const _Float16* p = base + (size_t)row * ld + col;
  v8h lo = *reinterpret_cast<const v8h*>(p);        // global_load_b128 / ds_load_b128
  v8h hi = *reinterpret_cast<const v8h*>(p + 16);
  v16h r;
#pragma unroll
  for (int i = 0; i < 8; ++i) { r[i] = lo[i]; r[i + 8] = hi[i]; }
  return r;
}

// Async global->LDS copy of 16 bytes per lane (gfx1250 VGLOBAL async path,
// tracked by ASYNCcnt). VDST VGPR carries the workgroup-relative LDS byte
// address (= low 32 bits of the generic LDS pointer).
__device__ __forceinline__ void async_copy16(const _Float16* gsrc, _Float16* ldst) {
  uint32_t laddr = (uint32_t)(uintptr_t)ldst;
  uint64_t gaddr = (uint64_t)(uintptr_t)gsrc;
  asm volatile("global_load_async_to_lds_b128 %0, %1, off"
               :: "v"(laddr), "v"(gaddr) : "memory");
}

__device__ __forceinline__ void wait_async0() {
  asm volatile("s_wait_asynccnt 0" ::: "memory");
}

// ---------------------------------------------------------------------------
// fp32 -> fp16 conversion (hidden states + weights), grid-stride
__global__ void cvt_f32_to_f16(const float* __restrict__ in,
                               _Float16* __restrict__ out, int n) {
  int i = blockIdx.x * blockDim.x + threadIdx.x;
  int stride = gridDim.x * blockDim.x;
  for (; i < n; i += stride) out[i] = (_Float16)in[i];
}

// ---------------------------------------------------------------------------
// QKV projection: out = hs @ W^T + bias, written in split-head layout.
//   transposed==0: out[b, head, s, d]  (Q, K)
//   transposed==1: out[b, head, d, s]  (V stored pre-transposed for P@V)
// One wave computes a 32(M) x 64(N) tile; N tile == one head.
__global__ void __launch_bounds__(256) qkv_proj(
    const _Float16* __restrict__ hs, const _Float16* __restrict__ W,
    const float* __restrict__ bias, _Float16* __restrict__ out, int transposed) {
  int lane = threadIdx.x & 31;
  int wave = threadIdx.x >> 5;
  int wid  = blockIdx.x * 8 + wave;      // 0..3071
  int mt = wid / NH_;                    // 0..255
  int nt = wid % NH_;                    // head, 0..11
  int m0 = mt * 32;
  int n0 = nt * HD_;

  v8f zero = {};
  v8f acc[2][4];
#pragma unroll
  for (int mi = 0; mi < 2; ++mi)
#pragma unroll
    for (int j = 0; j < 4; ++j) acc[mi][j] = zero;

  for (int kk = 0; kk < H_; kk += 32) {
    v16h a0 = load_frag(hs + (size_t)m0 * H_ + kk, H_, lane);
    v16h a1 = load_frag(hs + (size_t)(m0 + 16) * H_ + kk, H_, lane);
#pragma unroll
    for (int j = 0; j < 4; ++j) {
      v16h bf = load_frag(W + (size_t)(n0 + 16 * j) * H_ + kk, H_, lane);
      acc[0][j] = wmma16(a0, bf, acc[0][j]);
      acc[1][j] = wmma16(a1, bf, acc[1][j]);
    }
  }

  int nlo = lane & 15, half = lane >> 4;
  float bv[4];
#pragma unroll
  for (int j = 0; j < 4; ++j) bv[j] = bias[n0 + 16 * j + nlo];

#pragma unroll
  for (int mi = 0; mi < 2; ++mi)
#pragma unroll
    for (int j = 0; j < 4; ++j)
#pragma unroll
      for (int r = 0; r < 8; ++r) {
        int m = m0 + mi * 16 + r + 8 * half;       // global row in [0, M_)
        int d = 16 * j + nlo;                      // 0..63 within head
        int b = m >> 12;                           // m / S_
        int s = m & (S_ - 1);
        float v = acc[mi][j][r] + bv[j];
        size_t addr = transposed
            ? (((size_t)(b * NH_ + nt) * HD_ + d) * S_ + s)
            : (((size_t)(b * NH_ + nt) * S_ + s) * HD_ + d);
        out[addr] = (_Float16)v;
      }
}

// ---------------------------------------------------------------------------
// Attention. One block = (b, head, 128 query rows); 8 waves each own 16 rows.
// K / V^T tiles are staged cooperatively into LDS via async global->LDS copies
// (shared by all 8 waves, cutting global operand traffic 8x), then consumed by
// ds_load_b128 WMMA fragment loads.
// Pass 1: online per-lane softmax stats over all keys (QK^T via WMMA).
// Pass 2: recompute scores, write normalized probs once, stage P tiles in LDS,
//         accumulate ctx = P @ V via WMMA against pre-transposed V.
__global__ void __launch_bounds__(256) attn(
    const _Float16* __restrict__ q,    // [B,NH,S,HD]
    const _Float16* __restrict__ k,    // [B,NH,S,HD]
    const _Float16* __restrict__ vT,   // [B,NH,HD,S]
    const float* __restrict__ mask,    // [B,S]
    float* __restrict__ ctx,           // [B,S,H]
    float* __restrict__ probs) {       // [B,NH,S,S]
  __shared__ __align__(16) _Float16 ktile[KT_][HD_];     // 16 KB: 128 keys x 64 hd
  __shared__ __align__(16) _Float16 vtile[HD_][KT_];     // 16 KB: 64 d x 128 keys
  __shared__ __align__(16) _Float16 pstage[8][16][40];   // 10 KB per-wave P tiles

  int lane = threadIdx.x & 31;
  int wave = threadIdx.x >> 5;
  int tid  = threadIdx.x;
  int bh   = blockIdx.x >> 5;          // 0..23
  int qblk = blockIdx.x & 31;          // 0..31
  int b = bh / NH_, h = bh % NH_;
  int q0 = qblk * 128 + wave * 16;     // sequence-local query row base

  const _Float16* qb = q  + (size_t)bh * S_ * HD_;
  const _Float16* kb = k  + (size_t)bh * S_ * HD_;
  const _Float16* vb = vT + (size_t)bh * HD_ * S_;
  const float*    mb = mask + (size_t)b * S_;
  float*          pb = probs + (size_t)bh * S_ * S_;

  int nlo = lane & 15, half = lane >> 4;
  const float scale = 0.125f;          // 1/sqrt(HD)

  v16h qf0 = load_frag(qb + (size_t)q0 * HD_, HD_, lane);       // HD 0..31
  v16h qf1 = load_frag(qb + (size_t)q0 * HD_ + 32, HD_, lane);  // HD 32..63

  float rm[8], rs[8];
#pragma unroll
  for (int r = 0; r < 8; ++r) { rm[r] = -3.0e38f; rs[r] = 0.0f; }

  // ---- Pass 1: online max/sum over all keys, K tiles staged in LDS
  for (int kt0 = 0; kt0 < S_; kt0 += KT_) {
    const _Float16* gk = kb + (size_t)kt0 * HD_;   // contiguous 16 KB block
#pragma unroll
    for (int j = 0; j < 4; ++j) {
      int off = tid * 8 + j * 2048;                // halves
      async_copy16(gk + off, &ktile[0][0] + off);
    }
    wait_async0();
    __syncthreads();

    for (int kt = 0; kt < KT_; kt += 16) {
      v16h kf0 = load_frag(&ktile[kt][0], HD_, lane);
      v16h kf1 = load_frag(&ktile[kt][0] + 32, HD_, lane);
      v8f sc = {};
      sc = wmma16(qf0, kf0, sc);
      sc = wmma16(qf1, kf1, sc);
      float mv = mb[kt0 + kt + nlo];
#pragma unroll
      for (int r = 0; r < 8; ++r) {
        float s = sc[r] * scale + mv;
        float mn = fmaxf(rm[r], s);
        rs[r] = rs[r] * __expf(rm[r] - mn) + __expf(s - mn);
        rm[r] = mn;
      }
    }
    __syncthreads();   // tile consumed by all waves before next overwrite
  }

  // combine the 16 lanes of each half (row = r + 8*half)
#pragma unroll
  for (int r = 0; r < 8; ++r) {
#pragma unroll
    for (int off = 8; off >= 1; off >>= 1) {
      float om = __shfl_xor(rm[r], off, 16);
      float os = __shfl_xor(rs[r], off, 16);
      float nm = fmaxf(rm[r], om);
      rs[r] = rs[r] * __expf(rm[r] - nm) + os * __expf(om - nm);
      rm[r] = nm;
    }
  }
  float inv[8];
#pragma unroll
  for (int r = 0; r < 8; ++r) inv[r] = 1.0f / rs[r];

  v8f zero = {};
  v8f cacc[4];
#pragma unroll
  for (int j = 0; j < 4; ++j) cacc[j] = zero;

  _Float16* ps = &pstage[wave][0][0];

  // ---- Pass 2: recompute scores, emit probs once, accumulate ctx = P @ V
  for (int kt0 = 0; kt0 < S_; kt0 += KT_) {
    const _Float16* gk = kb + (size_t)kt0 * HD_;
#pragma unroll
    for (int j = 0; j < 4; ++j) {
      int off = tid * 8 + j * 2048;
      async_copy16(gk + off, &ktile[0][0] + off);
    }
    // V^T tile: 64 rows x 128 halves, global row stride S_
#pragma unroll
    for (int j = 0; j < 4; ++j) {
      int c = tid + j * 256;                       // chunk 0..1023
      int row = c >> 4;                            // 0..63
      int col = (c & 15) * 8;                      // halves 0..120
      async_copy16(vb + (size_t)row * S_ + kt0 + col, &vtile[row][col]);
    }
    wait_async0();
    __syncthreads();

    for (int kk = 0; kk < KT_; kk += 32) {
#pragma unroll
      for (int t = 0; t < 2; ++t) {
        int kt = kk + t * 16;
        v16h kf0 = load_frag(&ktile[kt][0], HD_, lane);
        v16h kf1 = load_frag(&ktile[kt][0] + 32, HD_, lane);
        v8f sc = {};
        sc = wmma16(qf0, kf0, sc);
        sc = wmma16(qf1, kf1, sc);
        float mv = mb[kt0 + kt + nlo];
#pragma unroll
        for (int r = 0; r < 8; ++r) {
          int m = r + 8 * half;
          float p = __expf(sc[r] * scale + mv - rm[r]) * inv[r];
          pb[(size_t)(q0 + m) * S_ + kt0 + kt + nlo] = p;   // the one probs write
          ps[m * 40 + t * 16 + nlo] = (_Float16)p;          // stage for P@V
        }
      }
      // DS ops are in-order per wave; wait for staging stores before reload.
      asm volatile("s_wait_dscnt 0" ::: "memory");
      v16h pf = load_frag(ps, 40, lane);                    // A frag (16q x 32k)
#pragma unroll
      for (int j = 0; j < 4; ++j) {
        v16h vf = load_frag(&vtile[16 * j][kk], KT_, lane);
        cacc[j] = wmma16(pf, vf, cacc[j]);
      }
    }
    __syncthreads();
  }

#pragma unroll
  for (int j = 0; j < 4; ++j)
#pragma unroll
    for (int r = 0; r < 8; ++r) {
      int m = r + 8 * half;
      int d = 16 * j + nlo;
      ctx[((size_t)b * S_ + q0 + m) * H_ + h * HD_ + d] = cacc[j][r];
    }
}

// ---------------------------------------------------------------------------
extern "C" void kernel_launch(void* const* d_in, const int* in_sizes, int n_in,
                              void* d_out, int out_size, void* d_ws, size_t ws_size,
                              hipStream_t stream) {
  const float* hs   = (const float*)d_in[0];
  const float* mask = (const float*)d_in[1];
  const float* Wq   = (const float*)d_in[2];
  const float* bq   = (const float*)d_in[3];
  const float* Wk   = (const float*)d_in[4];
  const float* bk   = (const float*)d_in[5];
  const float* Wv   = (const float*)d_in[6];
  const float* bv   = (const float*)d_in[7];

  float* ctx   = (float*)d_out;
  float* probs = ctx + (size_t)B_ * S_ * H_;   // outputs concatenated: ctx, probs

  // workspace carve-up (f16): hs16 | Wq16 | Wk16 | Wv16 | q16 | k16 | vT16
  _Float16* w = (_Float16*)d_ws;
  size_t o = 0;
  _Float16* hs16 = w + o; o += (size_t)M_ * H_;
  _Float16* wq16 = w + o; o += (size_t)H_ * H_;
  _Float16* wk16 = w + o; o += (size_t)H_ * H_;
  _Float16* wv16 = w + o; o += (size_t)H_ * H_;
  _Float16* q16  = w + o; o += (size_t)M_ * H_;
  _Float16* k16  = w + o; o += (size_t)M_ * H_;
  _Float16* vT16 = w + o; o += (size_t)M_ * H_;

  cvt_f32_to_f16<<<2048, 256, 0, stream>>>(hs, hs16, M_ * H_);
  cvt_f32_to_f16<<<1024, 256, 0, stream>>>(Wq, wq16, H_ * H_);
  cvt_f32_to_f16<<<1024, 256, 0, stream>>>(Wk, wk16, H_ * H_);
  cvt_f32_to_f16<<<1024, 256, 0, stream>>>(Wv, wv16, H_ * H_);

  // 3072 waves per projection: grid = 3072/8
  qkv_proj<<<384, 256, 0, stream>>>(hs16, wq16, bq, q16, 0);
  qkv_proj<<<384, 256, 0, stream>>>(hs16, wk16, bk, k16, 0);
  qkv_proj<<<384, 256, 0, stream>>>(hs16, wv16, bv, vT16, 1);

  // 24 (b,h) pairs x 32 query blocks of 128 rows
  attn<<<768, 256, 0, stream>>>(q16, k16, vT16, mask, ctx, probs);
}
